// CrossAttention_60979945668660
// MI455X (gfx1250) — compile-verified, matrix operand order checked by reading, not run on hardware
//
#include <hip/hip_runtime.h>

#define B_    4
#define L_    1024
#define H_    8
#define E_    64
#define WIN_  12
#define NB    25                       // 2*WIN+1 distance buckets
#define SL    0.18033688011112042f     // (1/sqrt(64)) * log2(e): scores in log2 units
#define TW    64                       // K-tile width (columns per iteration)

typedef __attribute__((ext_vector_type(16))) _Float16 v16h;
typedef __attribute__((ext_vector_type(8)))  _Float16 v8h;
typedef __attribute__((ext_vector_type(4)))  _Float16 h4;
typedef __attribute__((ext_vector_type(8)))  float    v8f;

// LDS row strides in halves; all chunk addresses stay 16B aligned for ds_read_b128
#define KT_S 72   // K tile: 64 rows x 64 halves
#define VT_S 72   // V tile transposed: 64 (e) x 64 (k) halves
#define PT_S 72   // per-wave probability tile: 16 x 64 halves
#define WT_S 40   // per-wave bucket-weight tile: 16 x 32 halves (25 + pad)
#define TB_S 40   // rel-value table transposed: 64 (e) x 32 (d) halves

static __device__ __forceinline__ v8f wmma16(v16h a, v16h b, v8f c) {
  return __builtin_amdgcn_wmma_f32_16x16x32_f16(false, a, false, b, (short)0, c,
                                                false, false);
}
static __device__ __forceinline__ float fexp2(float x) {
#if __has_builtin(__builtin_amdgcn_exp2f)
  return __builtin_amdgcn_exp2f(x);
#else
  return exp2f(x);
#endif
}
static __device__ __forceinline__ float frcp(float x) {
#if __has_builtin(__builtin_amdgcn_rcpf)
  return __builtin_amdgcn_rcpf(x);
#else
  return 1.0f / x;
#endif
}
// A/B fragment: half i<8 holds k=base+i, half i>=8 holds k=base+16+(i-8)
static __device__ __forceinline__ v16h ld16h(const _Float16* p0, const _Float16* p1) {
  v8h lo = *(const v8h*)p0;
  v8h hi = *(const v8h*)p1;
  v16h r;
#pragma unroll
  for (int i = 0; i < 8; ++i) { r[i] = lo[i]; r[i + 8] = hi[i]; }
  return r;
}
static __device__ __forceinline__ int iclamp(int v, int lo, int hi) {
  return v < lo ? lo : (v > hi ? hi : v);
}

// --------------------------------------------------------------------------
// Kernel 0: P[b,l,h,d] = sum_e x[b,l,h,e] * rel_k_table[d,e]
// --------------------------------------------------------------------------
__global__ __launch_bounds__(256) void proj_kernel(const float* __restrict__ x,
                                                   const float* __restrict__ table,
                                                   float* __restrict__ P) {
  __shared__ float tl[NB * E_];
  for (int i = threadIdx.x; i < NB * E_; i += 256) tl[i] = table[i];
  __syncthreads();
  int gid = blockIdx.x * 256 + threadIdx.x;
  const int total = B_ * L_ * H_ * NB;
  if (gid >= total) return;
  int d = gid % NB;
  int tok = gid / NB;
  const float* xr = x + tok * E_;
  float acc = 0.f;
#pragma unroll
  for (int e = 0; e < E_; ++e) acc += xr[e] * tl[d * E_ + e];
  P[gid] = acc;
}

// --------------------------------------------------------------------------
// Flash pass. passB==0: Q=x,K=y,V=v_y, bias=P[row,d]      -> V_Y
//             passB==1: Q=y,K=x,V=v_x, bias=P[col,24-d]   -> V_X
// Block = 128 threads (4 waves), owns 64 Q rows of one (b,h).
// Each wave: 16 rows, streams 64-wide K tiles (16 iterations).
// --------------------------------------------------------------------------
__global__ __launch_bounds__(128) void flash_kernel(
    const float* __restrict__ Q, const float* __restrict__ K,
    const float* __restrict__ V, const float* __restrict__ P,
    const float* __restrict__ tabV, float* __restrict__ out, int passB) {
  __shared__ __align__(16) _Float16 sK[TW * KT_S];
  __shared__ __align__(16) _Float16 sV[E_ * VT_S];      // [e][k]
  __shared__ __align__(16) _Float16 sPr[4][16 * PT_S];  // probabilities
  __shared__ __align__(16) _Float16 sPm[4][16 * PT_S];  // left-masked probs
  __shared__ __align__(16) _Float16 sWh[4][16 * WT_S];  // bucket weights (f16)
  __shared__ __align__(16) _Float16 sTb[E_ * TB_S];     // rel-value table [e][d]
  __shared__ float sBand[4][16][23];                    // band scores (log2 units)
  __shared__ float sBias[TW * NB];                      // P rows (prescaled by SL)
  __shared__ float sE0[TW], sE24[TW];                   // edge-bucket cols (passB)
  __shared__ float sZ[4][16], sAL[4][16];

  const int tid  = threadIdx.x;
  const int wv   = tid >> 5;
  const int lane = tid & 31;
  const int n    = lane & 15;
  const int ho   = (lane >> 4) << 3;  // 0 or 8
  const int bh = blockIdx.y;
  const int b = bh / H_, h = bh % H_;
  const int l0 = blockIdx.x * 64;
  const int lq = l0 + wv * 16;

  const int TOKS = H_ * E_;                     // token stride in Q/K/V/out
  const int TOKP = H_ * NB;                     // token stride in P
  const int kvbase  = (b * L_ * H_ + h) * E_;
  const int pbase   = (b * L_ * H_ + h) * NB;
  const int outbase = (b * L_ * H_ + h) * E_;

  // ---- one-time staging ----
  for (int i = tid; i < E_ * 32; i += 128) {    // rel-value table, transposed f16
    int e = i >> 5, k = i & 31;
    sTb[e * TB_S + k] = (_Float16)((k < NB) ? tabV[k * E_ + e] : 0.f);
  }
  if (!passB) {                                  // Q-row bias table (prescaled)
    for (int i = tid; i < 64 * NB; i += 128) {
      int r = i / NB, d = i % NB;
      sBias[i] = P[pbase + (l0 + r) * TOKP + d] * SL;
    }
  }
  for (int i = tid; i < 4 * 16 * 23; i += 128) (&sBand[0][0][0])[i] = -3.0e38f;
  __syncthreads();

  // ---- Q fragments, prescaled by SL ----
  const float* qp = Q + kvbase + (lq + n) * TOKS;
  v16h q0, q1;
#pragma unroll
  for (int i = 0; i < 8; ++i) {
    q0[i]     = (_Float16)(qp[ho + i] * SL);
    q0[i + 8] = (_Float16)(qp[ho + 16 + i] * SL);
    q1[i]     = (_Float16)(qp[ho + 32 + i] * SL);
    q1[i + 8] = (_Float16)(qp[ho + 48 + i] * SL);
  }
  v16h ones;
#pragma unroll
  for (int i = 0; i < 16; ++i) ones[i] = (_Float16)1.0f;

  // pass A fast-path per-row edge biases in registers
  float rb0[8], rb24[8];
#pragma unroll
  for (int v = 0; v < 8; ++v) { rb0[v] = 0.f; rb24[v] = 0.f; }
  if (!passB) {
#pragma unroll
    for (int v = 0; v < 8; ++v) {
      const float* pp = P + pbase + (lq + v + ho) * TOKP;
      rb0[v] = pp[0] * SL;
      rb24[v] = pp[24] * SL;
    }
  }

  v8f os[4] = {v8f{}, v8f{}, v8f{}, v8f{}};
  v8f oz = {}, ol = {};
  float m = -3.0e38f;

  for (int it = 0; it < L_ / TW; ++it) {
    const int s0 = it * TW;
    const bool blockSlow = (s0 >= l0 - 75) && (s0 <= l0 + 74);
    __syncthreads();
    // ---- cooperative staging (K row-major + V transposed), f32 -> f16 ----
    for (int i = tid; i < TW * 16; i += 128) {
      int sl = i >> 4, c4 = (i & 15) << 2;
      const float* rowp = K + kvbase + (s0 + sl) * TOKS;
      float4 fk = *(const float4*)(rowp + c4);
      h4 r;
      r[0] = (_Float16)fk.x; r[1] = (_Float16)fk.y;
      r[2] = (_Float16)fk.z; r[3] = (_Float16)fk.w;
      *(h4*)&sK[sl * KT_S + c4] = r;
      const float* rowv = V + kvbase + (s0 + sl) * TOKS;
      float4 fv = *(const float4*)(rowv + c4);
      sV[(c4 + 0) * VT_S + sl] = (_Float16)fv.x;
      sV[(c4 + 1) * VT_S + sl] = (_Float16)fv.y;
      sV[(c4 + 2) * VT_S + sl] = (_Float16)fv.z;
      sV[(c4 + 3) * VT_S + sl] = (_Float16)fv.w;
    }
    if (passB) {
      if (tid < TW) {                                // edge-bucket bias columns
        const float* pp = P + pbase + (s0 + tid) * TOKP;
        sE0[tid] = pp[0] * SL;
        sE24[tid] = pp[24] * SL;
      }
      if (blockSlow) {                               // full bias rows (rare)
        for (int i = tid; i < TW * NB; i += 128) {
          int sl = i / NB, d = i % NB;
          sBias[i] = P[pbase + (s0 + sl) * TOKP + d] * SL;
        }
      }
    }
    if (s0 + TW < L_) {                              // prefetch next tile
      int r = tid >> 1, half = (tid & 1) << 5;
      __builtin_prefetch(K + kvbase + (s0 + TW + r) * TOKS + half, 0, 3);
      __builtin_prefetch(V + kvbase + (s0 + TW + r) * TOKS + half, 0, 3);
    }
    __syncthreads();

    // ---- scores: preload all 8 B-fragments, then 8 WMMAs ----
    v16h kf[8];
#pragma unroll
    for (int g = 0; g < 4; ++g) {
      const _Float16* kp = sK + (16 * g + n) * KT_S;
      kf[2 * g]     = ld16h(kp + ho, kp + ho + 16);
      kf[2 * g + 1] = ld16h(kp + ho + 32, kp + ho + 48);
    }
    v8f cs[4];
#pragma unroll
    for (int g = 0; g < 4; ++g) {
      v8f c = {};
      c = wmma16(q0, kf[2 * g], c);
      c = wmma16(q1, kf[2 * g + 1], c);
      cs[g] = c;
    }

    // ---- bias: wave-uniform fast/slow classification ----
    const bool farLeft  = (s0 <= lq - 75);   // all deltas <= -12
    const bool farRight = (s0 >= lq + 27);   // all deltas >= +12
    const bool slowT = !(farLeft || farRight);
    if (slowT) {
#pragma unroll
      for (int g = 0; g < 4; ++g) {
        int scol = s0 + 16 * g + n;
#pragma unroll
        for (int v = 0; v < 8; ++v) {
          int mr = v + ho, l = lq + mr, dd = scol - l;
          int bi = iclamp(dd, -WIN_, WIN_) + WIN_;
          float bias = passB ? sBias[(16 * g + n) * NB + (24 - bi)]
                             : sBias[(wv * 16 + mr) * NB + bi];
          float sc = cs[g][v] + bias;
          if (dd >= -11 && dd <= 11) sBand[wv][mr][dd + 11] = sc;
          cs[g][v] = sc;
        }
      }
    } else if (passB) {
      const float* e = farLeft ? sE24 : sE0;   // d=0 -> P[s,24]; d=24 -> P[s,0]
      float b0 = e[n], b1 = e[16 + n], b2 = e[32 + n], b3 = e[48 + n];
#pragma unroll
      for (int v = 0; v < 8; ++v) {
        cs[0][v] += b0; cs[1][v] += b1; cs[2][v] += b2; cs[3][v] += b3;
      }
    } else {
#pragma unroll
      for (int v = 0; v < 8; ++v) {
        float rb = farLeft ? rb0[v] : rb24[v];
        cs[0][v] += rb; cs[1][v] += rb; cs[2][v] += rb; cs[3][v] += rb;
      }
    }

    // ---- wave-level running max (one 5-step butterfly per tile) ----
    float tm = cs[0][0];
#pragma unroll
    for (int g = 0; g < 4; ++g)
#pragma unroll
      for (int v = 0; v < 8; ++v) tm = fmaxf(tm, cs[g][v]);
    tm = fmaxf(tm, __shfl_xor(tm, 1, 32));
    tm = fmaxf(tm, __shfl_xor(tm, 2, 32));
    tm = fmaxf(tm, __shfl_xor(tm, 4, 32));
    tm = fmaxf(tm, __shfl_xor(tm, 8, 32));
    tm = fmaxf(tm, __shfl_xor(tm, 16, 32));
    float nm = fmaxf(m, tm);
    int upd = (nm > m) ? 1 : 0;
    if (__builtin_amdgcn_readfirstlane(upd)) {     // uniform: skip once stable
      float f = fexp2(m - nm);
#pragma unroll
      for (int g = 0; g < 4; ++g)
#pragma unroll
        for (int v = 0; v < 8; ++v) os[g][v] *= f;
#pragma unroll
      for (int v = 0; v < 8; ++v) { oz[v] *= f; ol[v] *= f; }
      m = nm;
    }

    // ---- preload V fragments early (independent of probability stores) ----
    v16h bv[8];
#pragma unroll
    for (int g = 0; g < 4; ++g) {
      const _Float16* vb = sV + (16 * g + n) * VT_S;
      bv[2 * g]     = ld16h(vb + ho, vb + ho + 16);
      bv[2 * g + 1] = ld16h(vb + ho + 32, vb + ho + 48);
    }

    // ---- probabilities (exp2), C->A relayout via per-wave LDS ----
#pragma unroll
    for (int g = 0; g < 4; ++g) {
      int scol = s0 + 16 * g + n;
#pragma unroll
      for (int v = 0; v < 8; ++v) {
        int mr = v + ho;
        float p = fexp2(cs[g][v] - m);
        sPr[wv][mr * PT_S + 16 * g + n] = (_Float16)p;
        if (slowT) {
          float pm = ((scol - (lq + mr)) <= -WIN_) ? p : 0.f;
          sPm[wv][mr * PT_S + 16 * g + n] = (_Float16)pm;
        }
      }
    }
    const _Float16* pb = &sPr[wv][n * PT_S];
    v16h pa0 = ld16h(pb + ho, pb + ho + 16);
    v16h pa1 = ld16h(pb + ho + 32, pb + ho + 48);

    // Z (row sums) and left-tail mass via ones-column WMMAs
    oz = wmma16(pa0, ones, oz);
    oz = wmma16(pa1, ones, oz);
    if (farLeft) {
      ol = wmma16(pa0, ones, ol);
      ol = wmma16(pa1, ones, ol);
    } else if (slowT) {
      const _Float16* pm = &sPm[wv][n * PT_S];
      ol = wmma16(ld16h(pm + ho, pm + ho + 16), ones, ol);
      ol = wmma16(ld16h(pm + ho + 32, pm + ho + 48), ones, ol);
    }
    // O += p @ V : 8 WMMAs on preloaded fragments
#pragma unroll
    for (int g = 0; g < 4; ++g) {
      os[g] = wmma16(pa0, bv[2 * g], os[g]);
      os[g] = wmma16(pa1, bv[2 * g + 1], os[g]);
    }
  }

  // ---- epilogue: every lane already holds its row's Z (oz) and left mass ----
  if (n == 0) {
#pragma unroll
    for (int v = 0; v < 8; ++v) {
      sZ[wv][v + ho] = oz[v];
      sAL[wv][v + ho] = ol[v];
    }
  }
  if (lane < 16) {   // per-row bucket weights -> f16 A-layout tile
    int r = lane;
    float inv = frcp(sZ[wv][r]);
    float wl = sAL[wv][r] * inv;
    float acc = wl;
    _Float16* wrow = &sWh[wv][r * WT_S];
    wrow[0] = (_Float16)wl;
#pragma unroll
    for (int j = 0; j < 23; ++j) {
      float pb2 = fexp2(sBand[wv][r][j] - m) * inv;
      wrow[1 + j] = (_Float16)pb2;
      acc += pb2;
    }
    wrow[24] = (_Float16)fmaxf(1.0f - acc, 0.0f);
#pragma unroll
    for (int k = 25; k < 32; ++k) wrow[k] = (_Float16)0.0f;
  }

  float iz[8];
#pragma unroll
  for (int v = 0; v < 8; ++v) iz[v] = frcp(oz[v]);
  const _Float16* wb = &sWh[wv][n * WT_S];
  v16h paw = ld16h(wb + ho, wb + ho + 16);
#pragma unroll
  for (int g = 0; g < 4; ++g) {
    v8f cg;
#pragma unroll
    for (int v = 0; v < 8; ++v) cg[v] = os[g][v] * iz[v];   // normalize
    const _Float16* tb = sTb + (16 * g + n) * TB_S;
    cg = wmma16(paw, ld16h(tb + ho, tb + ho + 16), cg);     // + W @ table
#pragma unroll
    for (int v = 0; v < 8; ++v)
      out[outbase + (lq + v + ho) * TOKS + 16 * g + n] = cg[v];
  }
}

// --------------------------------------------------------------------------
extern "C" void kernel_launch(void* const* d_in, const int* in_sizes, int n_in,
                              void* d_out, int out_size, void* d_ws, size_t ws_size,
                              hipStream_t stream) {
  (void)in_sizes; (void)n_in; (void)out_size; (void)ws_size;
  const float* x      = (const float*)d_in[0];
  const float* y      = (const float*)d_in[1];
  const float* v_x    = (const float*)d_in[2];
  const float* v_y    = (const float*)d_in[3];
  const float* rel_k  = (const float*)d_in[4];
  const float* rel_vx = (const float*)d_in[5];
  const float* rel_vy = (const float*)d_in[6];

  float* P = (float*)d_ws;  // B*L*H*25 floats = 3.3 MB
  float* outVY = (float*)d_out;
  float* outVX = outVY + (size_t)B_ * L_ * H_ * E_;

  const int totP = B_ * L_ * H_ * NB;
  proj_kernel<<<(totP + 255) / 256, 256, 0, stream>>>(x, rel_k, P);

  dim3 grid(L_ / 64, B_ * H_);
  flash_kernel<<<grid, 128, 0, stream>>>(x, y, v_y, P, rel_vy, outVY, 0);
  flash_kernel<<<grid, 128, 0, stream>>>(y, x, v_x, P, rel_vx, outVX, 1);
}
